// SwinTransformerBlock_15564961481374
// MI455X (gfx1250) — matrix operand order repaired and checked by reading, not compiled, not run on hardware
//
#include <hip/hip_runtime.h>

typedef unsigned short u16;
typedef __attribute__((ext_vector_type(16))) __bf16 v16bf;
typedef __attribute__((ext_vector_type(8)))  float  v8f;

#define C_    512
#define NH_   16
#define SCALE_ 0.17677669529663687f

// LDS staging geometry: A tile 64x32 (rows padded to 40), B tile 128x32 (padded 40)
#define APAD   40
#define ATILE  (64 * APAD)            // 2560 u16
#define BTILE  (128 * APAD)           // 5120 u16
#define BUFSZ  (ATILE + BTILE)        // 7680 u16 per buffer
#define SMEM_BYTES (2 * BUFSZ * 2)    // 30720 bytes (double buffered)

// ---------- helpers ----------
__device__ __forceinline__ u16 f2bf(float f) {
    unsigned int u = __float_as_uint(f);
    u += 0x7FFFu + ((u >> 16) & 1u);            // round to nearest even
    return (u16)(u >> 16);
}

// A-fragment (16x32, MxK): lane row M=lane&15, K halves {koff..koff+7, 16+koff..}
__device__ __forceinline__ v16bf ldfragA(const u16* p) {
    union { v16bf v; uint4 u[2]; } t;
    t.u[0] = *(const uint4*)(p);
    t.u[1] = *(const uint4*)(p + 16);
    return t.v;
}
// B-fragment (32x16, KxN): lane n holds K halves {koff..koff+7, 16+koff..}
__device__ __forceinline__ v16bf ldfragB(const u16* p) {
    union { v16bf v; uint4 u[2]; } t;
    t.u[0] = *(const uint4*)(p);
    t.u[1] = *(const uint4*)(p + 16);
    return t.v;
}
__device__ __forceinline__ v8f wmma_bf16(v16bf a, v16bf b, v8f c) {
    return __builtin_amdgcn_wmma_f32_16x16x32_bf16(false, a, false, b, (short)0, c, false, false);
}

// CDNA5 async copy: 16B per lane, global -> LDS, tracked by ASYNCcnt
__device__ __forceinline__ void async_cp16(unsigned lds_byte_addr, const u16* gsrc) {
    asm volatile("global_load_async_to_lds_b128 %0, %1, off"
                 :: "v"(lds_byte_addr), "v"((unsigned long long)gsrc)
                 : "memory");
}
__device__ __forceinline__ void wait_async_le3() {
    asm volatile("s_wait_asynccnt 0x3" ::: "memory");
}
__device__ __forceinline__ void wait_async_0() {
    asm volatile("s_wait_asynccnt 0x0" ::: "memory");
}

// ---------- weight cast + transpose: dst[n*K+k] = bf16(src[k*Nt+n]) ----------
__global__ void castT_kernel(const float* __restrict__ src, u16* __restrict__ dst,
                             int K, int Nt) {
    int i = blockIdx.x * 256 + threadIdx.x;     // i in [0, K*Nt)
    int n = i / K;
    int k = i - n * K;
    dst[i] = f2bf(src[(size_t)k * Nt + n]);
}

// ---------- LN1 + shift + window partition -> bf16 [BW, 64, 512] ----------
__global__ void ln1_win_kernel(const float* __restrict__ x, const float* __restrict__ g,
                               const float* __restrict__ b, u16* __restrict__ xw) {
    int lane = threadIdx.x & 31, wid = threadIdx.x >> 5;
    int tk = blockIdx.x * 8 + wid;              // destination windowed token
    int w  = tk >> 6, t = tk & 63;
    int bb = w >> 6, wi = w & 63;
    int hh = ((wi >> 3) << 3) + (t >> 3);
    int ww = ((wi & 7) << 3) + (t & 7);
    int sh = (hh + 4) & 63, sw = (ww + 4) & 63; // roll(-SHIFT) gather
    const float4* row = (const float4*)(x + ((size_t)bb * 4096 + sh * 64 + sw) * C_);
    float4 f[4];
    float s = 0.f, ss = 0.f;
    #pragma unroll
    for (int j = 0; j < 4; ++j) {
        f[j] = row[lane * 4 + j];
        s  += f[j].x + f[j].y + f[j].z + f[j].w;
        ss += f[j].x*f[j].x + f[j].y*f[j].y + f[j].z*f[j].z + f[j].w*f[j].w;
    }
    #pragma unroll
    for (int d = 16; d > 0; d >>= 1) { s += __shfl_xor(s, d, 32); ss += __shfl_xor(ss, d, 32); }
    float mu = s * (1.f / C_);
    float rs = rsqrtf(ss * (1.f / C_) - mu * mu + 1e-5f);
    unsigned int pr[8];
    #pragma unroll
    for (int j = 0; j < 4; ++j) {
        float4 g4 = ((const float4*)g)[lane * 4 + j];
        float4 b4 = ((const float4*)b)[lane * 4 + j];
        float a0 = (f[j].x - mu) * rs * g4.x + b4.x;
        float a1 = (f[j].y - mu) * rs * g4.y + b4.y;
        float a2 = (f[j].z - mu) * rs * g4.z + b4.z;
        float a3 = (f[j].w - mu) * rs * g4.w + b4.w;
        pr[2*j]   = (unsigned int)f2bf(a0) | ((unsigned int)f2bf(a1) << 16);
        pr[2*j+1] = (unsigned int)f2bf(a2) | ((unsigned int)f2bf(a3) << 16);
    }
    u16* dst = xw + (size_t)tk * C_ + lane * 16;
    *(uint4*)(dst)     = make_uint4(pr[0], pr[1], pr[2], pr[3]);
    *(uint4*)(dst + 8) = make_uint4(pr[4], pr[5], pr[6], pr[7]);
}

// ---------- LN2 -> bf16 [M, 512] ----------
__global__ void ln2_kernel(const float* __restrict__ x1, const float* __restrict__ g,
                           const float* __restrict__ b, u16* __restrict__ h2) {
    int lane = threadIdx.x & 31, wid = threadIdx.x >> 5;
    int tk = blockIdx.x * 8 + wid;
    const float4* row = (const float4*)(x1 + (size_t)tk * C_);
    float4 f[4];
    float s = 0.f, ss = 0.f;
    #pragma unroll
    for (int j = 0; j < 4; ++j) {
        f[j] = row[lane * 4 + j];
        s  += f[j].x + f[j].y + f[j].z + f[j].w;
        ss += f[j].x*f[j].x + f[j].y*f[j].y + f[j].z*f[j].z + f[j].w*f[j].w;
    }
    #pragma unroll
    for (int d = 16; d > 0; d >>= 1) { s += __shfl_xor(s, d, 32); ss += __shfl_xor(ss, d, 32); }
    float mu = s * (1.f / C_);
    float rs = rsqrtf(ss * (1.f / C_) - mu * mu + 1e-5f);
    unsigned int pr[8];
    #pragma unroll
    for (int j = 0; j < 4; ++j) {
        float4 g4 = ((const float4*)g)[lane * 4 + j];
        float4 b4 = ((const float4*)b)[lane * 4 + j];
        float a0 = (f[j].x - mu) * rs * g4.x + b4.x;
        float a1 = (f[j].y - mu) * rs * g4.y + b4.y;
        float a2 = (f[j].z - mu) * rs * g4.z + b4.z;
        float a3 = (f[j].w - mu) * rs * g4.w + b4.w;
        pr[2*j]   = (unsigned int)f2bf(a0) | ((unsigned int)f2bf(a1) << 16);
        pr[2*j+1] = (unsigned int)f2bf(a2) | ((unsigned int)f2bf(a3) << 16);
    }
    u16* dst = h2 + (size_t)tk * C_ + lane * 16;
    *(uint4*)(dst)     = make_uint4(pr[0], pr[1], pr[2], pr[3]);
    *(uint4*)(dst + 8) = make_uint4(pr[4], pr[5], pr[6], pr[7]);
}

// ---------- staged GEMM mainloop (async global->LDS, double buffered) ----------
// Block tile 64x128, 8 waves (2x4), each wave a 32x32 tile, K-chunks of 32.
__device__ __forceinline__ void stage_chunk(const u16* __restrict__ A, const u16* __restrict__ Bt,
                                            int K, int mB, int nB, int k,
                                            unsigned buf_byte, int tid) {
    // A tile: 64 rows x 4 x 16B segs, one async b128 per lane
    int arow = tid >> 2, aseg = tid & 3;
    async_cp16(buf_byte + (unsigned)(arow * APAD + aseg * 8) * 2,
               A + (size_t)(mB + arow) * K + k + aseg * 8);
    // B tile: 128 rows x 4 segs, two async b128 per lane
    #pragma unroll
    for (int r = 0; r < 2; ++r) {
        int i = tid + r * 256;
        int brow = i >> 2, bseg = i & 3;
        async_cp16(buf_byte + (unsigned)(ATILE + brow * APAD + bseg * 8) * 2,
                   Bt + (size_t)(nB + brow) * K + k + bseg * 8);
    }
}

__device__ __forceinline__ void gemm_loop_staged(const u16* __restrict__ A, const u16* __restrict__ Bt,
                                                 int K, int mB, int nB,
                                                 int wm, int wn, int lane, v8f acc[2][2]) {
    extern __shared__ u16 smem[];   // dynamic LDS, offset 0
    const int tid = threadIdx.x;
    const int mr = lane & 15;
    const int koffA = (lane >> 4) << 3;
    const int koffB = (lane >> 4) << 3;
    unsigned bufB[2] = { 0u, (unsigned)(BUFSZ * 2) };
    int cur = 0;
    stage_chunk(A, Bt, K, mB, nB, 0, bufB[0], tid);
    for (int k = 0; k < K; k += 32) {
        bool more = (k + 32) < K;
        if (more) {
            stage_chunk(A, Bt, K, mB, nB, k + 32, bufB[cur ^ 1], tid);
            wait_async_le3();       // previous chunk's 3 async ops done
        } else {
            wait_async_0();
        }
        __syncthreads();            // LDS from all waves visible
        const u16* sA = smem + (size_t)cur * BUFSZ;
        const u16* sB = sA + ATILE;
        v16bf af0 = ldfragA(sA + (wm * 32      + mr) * APAD + koffA);
        v16bf af1 = ldfragA(sA + (wm * 32 + 16 + mr) * APAD + koffA);
        v16bf bf0 = ldfragB(sB + (wn * 32      + mr) * APAD + koffB);
        v16bf bf1 = ldfragB(sB + (wn * 32 + 16 + mr) * APAD + koffB);
        acc[0][0] = wmma_bf16(af0, bf0, acc[0][0]);
        acc[0][1] = wmma_bf16(af0, bf1, acc[0][1]);
        acc[1][0] = wmma_bf16(af1, bf0, acc[1][0]);
        acc[1][1] = wmma_bf16(af1, bf1, acc[1][1]);
        __syncthreads();            // done reading before buffer reuse
        cur ^= 1;
    }
}

#define GEMM_PROLOG(KVAL)                                                   \
    const int lane = threadIdx.x & 31;                                      \
    const int wid  = threadIdx.x >> 5;                                      \
    const int m0 = blockIdx.y * 64 + (wid >> 2) * 32;                       \
    const int n0 = blockIdx.x * 128 + (wid & 3) * 32;                       \
    v8f acc[2][2] = {};                                                     \
    gemm_loop_staged(A, Bt, (KVAL), blockIdx.y * 64, blockIdx.x * 128,      \
                     (wid >> 2), (wid & 3), lane, acc);                     \
    const int half8 = (lane >> 4) << 3;

// ---------- QKV GEMM: scatter into q / k / v^T bf16 buffers ----------
__global__ __launch_bounds__(256) void qkv_gemm(const u16* __restrict__ A, const u16* __restrict__ Bt,
                                                const float* __restrict__ bias,
                                                u16* __restrict__ qb, u16* __restrict__ kb,
                                                u16* __restrict__ vb) {
    GEMM_PROLOG(512)
    #pragma unroll
    for (int mt = 0; mt < 2; ++mt)
    #pragma unroll
    for (int nt = 0; nt < 2; ++nt)
    #pragma unroll
    for (int r = 0; r < 8; ++r) {
        int m = m0 + mt * 16 + r + half8;
        int j = n0 + nt * 16 + (lane & 15);
        float v = acc[mt][nt][r] + bias[j];
        int which = j >> 9;
        int head  = (j >> 5) & 15;
        int hd    = j & 31;
        int w = m >> 6, t = m & 63;
        int gh = w * NH_ + head;
        if (which == 0)      qb[((size_t)gh * 64 + t) * 32 + hd] = f2bf(v * SCALE_);
        else if (which == 1) kb[((size_t)gh * 64 + t) * 32 + hd] = f2bf(v);
        else                 vb[((size_t)gh * 32 + hd) * 64 + t] = f2bf(v);  // v^T
    }
}

// ---------- attention: one wave per (window, head) ----------
__global__ __launch_bounds__(256) void attn_kernel(const u16* __restrict__ qb,
                                                   const u16* __restrict__ kb,
                                                   const u16* __restrict__ vb,
                                                   const float* __restrict__ rel_bias,
                                                   u16* __restrict__ attn_out) {
    __shared__ float sbias[225 * 16];
    __shared__ __align__(16) u16 sp[8][16][72];
    int tid = threadIdx.x;
    for (int i = tid; i < 225 * 16; i += 256) sbias[i] = rel_bias[i];
    __syncthreads();

    int lane = tid & 31, wid = tid >> 5;
    int gh = blockIdx.x * 8 + wid;
    int w = gh >> 4, head = gh & 15, wi = w & 63;
    int mr = lane & 15;
    int koffA = (lane >> 4) << 3;
    int koffB = (lane >> 4) << 3;
    int half8 = (lane >> 4) << 3;
    const u16* qbase = qb + (size_t)gh * 2048;
    const u16* kbase = kb + (size_t)gh * 2048;
    const u16* vbase = vb + (size_t)gh * 2048;
    const int whh = (wi >> 3) << 3, www = (wi & 7) << 3;

    for (int mi = 0; mi < 4; ++mi) {
        int q0 = mi * 16;
        v16bf aq = ldfragA(qbase + (q0 + mr) * 32 + koffA);
        v8f s[4];
        s[0] = s[1] = s[2] = s[3] = v8f{0.f,0.f,0.f,0.f,0.f,0.f,0.f,0.f};
        #pragma unroll
        for (int nt = 0; nt < 4; ++nt) {
            v16bf bk = ldfragB(kbase + (nt * 16 + mr) * 32 + koffB);
            s[nt] = wmma_bf16(aq, bk, s[nt]);
        }
        // bias + mask
        #pragma unroll
        for (int nt = 0; nt < 4; ++nt)
        #pragma unroll
        for (int r = 0; r < 8; ++r) {
            int qrow = q0 + r + half8;
            int kcol = nt * 16 + mr;
            int qy = qrow >> 3, qx = qrow & 7, ky = kcol >> 3, kx = kcol & 7;
            int idx = (qy - ky + 7) * 15 + (qx - kx + 7);
            float bv = sbias[idx * 16 + head];
            int qhh = whh + qy, qww = www + qx, khh = whh + ky, kww = www + kx;
            int rq = (qhh < 56 ? 0 : (qhh < 60 ? 1 : 2)) * 3 + (qww < 56 ? 0 : (qww < 60 ? 1 : 2));
            int rk = (khh < 56 ? 0 : (khh < 60 ? 1 : 2)) * 3 + (kww < 56 ? 0 : (kww < 60 ? 1 : 2));
            s[nt][r] += bv + ((rq != rk) ? -100.0f : 0.0f);
        }
        // softmax over 64 cols (16 lanes x 4 tiles)
        float mx[8], inv[8];
        #pragma unroll
        for (int r = 0; r < 8; ++r) {
            float m1 = fmaxf(fmaxf(s[0][r], s[1][r]), fmaxf(s[2][r], s[3][r]));
            #pragma unroll
            for (int d = 8; d > 0; d >>= 1) m1 = fmaxf(m1, __shfl_xor(m1, d, 16));
            mx[r] = m1;
        }
        #pragma unroll
        for (int nt = 0; nt < 4; ++nt)
        #pragma unroll
        for (int r = 0; r < 8; ++r) s[nt][r] = __expf(s[nt][r] - mx[r]);
        #pragma unroll
        for (int r = 0; r < 8; ++r) {
            float t1 = s[0][r] + s[1][r] + s[2][r] + s[3][r];
            #pragma unroll
            for (int d = 8; d > 0; d >>= 1) t1 += __shfl_xor(t1, d, 16);
            inv[r] = 1.0f / t1;
        }
        // normalized P -> LDS (bf16), re-layout for A-fragment reads
        #pragma unroll
        for (int nt = 0; nt < 4; ++nt)
        #pragma unroll
        for (int r = 0; r < 8; ++r)
            sp[wid][r + half8][nt * 16 + mr] = f2bf(s[nt][r] * inv[r]);
        // O = P @ V  (K = 64, two steps)
        v8f o[2];
        o[0] = o[1] = v8f{0.f,0.f,0.f,0.f,0.f,0.f,0.f,0.f};
        #pragma unroll
        for (int ks = 0; ks < 64; ks += 32) {
            v16bf ap = ldfragA(&sp[wid][mr][ks + koffA]);
            #pragma unroll
            for (int nt = 0; nt < 2; ++nt) {
                v16bf bv2 = ldfragB(vbase + (nt * 16 + mr) * 64 + ks + koffB);
                o[nt] = wmma_bf16(ap, bv2, o[nt]);
            }
        }
        #pragma unroll
        for (int nt = 0; nt < 2; ++nt)
        #pragma unroll
        for (int r = 0; r < 8; ++r) {
            int row = w * 64 + q0 + r + half8;
            int col = head * 32 + nt * 16 + mr;
            attn_out[(size_t)row * C_ + col] = f2bf(o[nt][r]);
        }
    }
}

// ---------- proj GEMM + window-reverse + un-shift + residual (fp32 x1) ----------
__global__ __launch_bounds__(256) void proj_gemm(const u16* __restrict__ A, const u16* __restrict__ Bt,
                                                 const float* __restrict__ bias,
                                                 const float* __restrict__ x,
                                                 float* __restrict__ x1) {
    GEMM_PROLOG(512)
    #pragma unroll
    for (int mt = 0; mt < 2; ++mt)
    #pragma unroll
    for (int nt = 0; nt < 2; ++nt)
    #pragma unroll
    for (int r = 0; r < 8; ++r) {
        int m = m0 + mt * 16 + r + half8;
        int j = n0 + nt * 16 + (lane & 15);
        float v = acc[mt][nt][r] + bias[j];
        int w = m >> 6, t = m & 63;
        int bb = w >> 6, wi = w & 63;
        int hh = ((wi >> 3) << 3) + (t >> 3);
        int ww = ((wi & 7) << 3) + (t & 7);
        int h  = (hh + 4) & 63, wc = (ww + 4) & 63;   // roll(+SHIFT)
        size_t idx = ((size_t)bb * 4096 + h * 64 + wc) * C_ + j;
        x1[idx] = x[idx] + v;
    }
}

// ---------- FC1 GEMM + exact GELU -> bf16 ----------
__global__ __launch_bounds__(256) void fc1_gemm(const u16* __restrict__ A, const u16* __restrict__ Bt,
                                                const float* __restrict__ bias,
                                                u16* __restrict__ out) {
    GEMM_PROLOG(512)
    #pragma unroll
    for (int mt = 0; mt < 2; ++mt)
    #pragma unroll
    for (int nt = 0; nt < 2; ++nt)
    #pragma unroll
    for (int r = 0; r < 8; ++r) {
        int m = m0 + mt * 16 + r + half8;
        int j = n0 + nt * 16 + (lane & 15);
        float v = acc[mt][nt][r] + bias[j];
        float g = 0.5f * v * (1.0f + erff(v * 0.7071067811865476f));
        out[(size_t)m * 2048 + j] = f2bf(g);
    }
}

// ---------- FC2 GEMM + residual -> fp32 output ----------
__global__ __launch_bounds__(256) void fc2_gemm(const u16* __restrict__ A, const u16* __restrict__ Bt,
                                                const float* __restrict__ bias,
                                                const float* __restrict__ x1,
                                                float* __restrict__ out) {
    GEMM_PROLOG(2048)
    #pragma unroll
    for (int mt = 0; mt < 2; ++mt)
    #pragma unroll
    for (int nt = 0; nt < 2; ++nt)
    #pragma unroll
    for (int r = 0; r < 8; ++r) {
        int m = m0 + mt * 16 + r + half8;
        int j = n0 + nt * 16 + (lane & 15);
        float v = acc[mt][nt][r] + bias[j];
        size_t idx = (size_t)m * C_ + j;
        out[idx] = x1[idx] + v;
    }
}

// ---------- host launch ----------
extern "C" void kernel_launch(void* const* d_in, const int* in_sizes, int n_in,
                              void* d_out, int out_size, void* d_ws, size_t ws_size,
                              hipStream_t stream) {
    (void)in_sizes; (void)n_in; (void)out_size; (void)ws_size;
    const float* x      = (const float*)d_in[0];
    const float* n1g    = (const float*)d_in[1];
    const float* n1b    = (const float*)d_in[2];
    const float* qkv_w  = (const float*)d_in[3];
    const float* qkv_b  = (const float*)d_in[4];
    const float* relb   = (const float*)d_in[5];
    const float* proj_w = (const float*)d_in[6];
    const float* proj_b = (const float*)d_in[7];
    const float* n2g    = (const float*)d_in[8];
    const float* n2b    = (const float*)d_in[9];
    const float* fc1w   = (const float*)d_in[10];
    const float* fc1b   = (const float*)d_in[11];
    const float* fc2w   = (const float*)d_in[12];
    const float* fc2b   = (const float*)d_in[13];

    char* ws = (char*)d_ws;
    const size_t MB = 1ull << 20;
    u16*   xw    = (u16*)(ws);                 // 64 MB, bf16 windowed LN1 output
    u16*   qb    = (u16*)(ws + 64 * MB);       // 64 MB
    u16*   kb    = (u16*)(ws + 128 * MB);      // 64 MB
    u16*   vb    = (u16*)(ws + 192 * MB);      // 64 MB (transposed)
    u16*   attn  = (u16*)(ws + 256 * MB);      // 64 MB
    float* x1    = (float*)(ws + 320 * MB);    // 128 MB fp32 residual
    u16*   h2    = (u16*)(ws + 256 * MB);      // reuses attn region after proj
    u16*   wqkvT = (u16*)(ws + 448 * MB);
    u16*   wprjT = (u16*)(ws + 452 * MB);
    u16*   wf1T  = (u16*)(ws + 456 * MB);
    u16*   wf2T  = (u16*)(ws + 460 * MB);
    u16*   fc1o  = (u16*)(ws);                 // 256 MB, reuses xw+qkv region

    castT_kernel<<<3072, 256, 0, stream>>>(qkv_w,  wqkvT, 512, 1536);
    castT_kernel<<<1024, 256, 0, stream>>>(proj_w, wprjT, 512, 512);
    castT_kernel<<<4096, 256, 0, stream>>>(fc1w,   wf1T,  512, 2048);
    castT_kernel<<<4096, 256, 0, stream>>>(fc2w,   wf2T,  2048, 512);

    ln1_win_kernel<<<8192, 256, 0, stream>>>(x, n1g, n1b, xw);
    qkv_gemm<<<dim3(12, 1024), 256, SMEM_BYTES, stream>>>(xw, wqkvT, qkv_b, qb, kb, vb);
    attn_kernel<<<2048, 256, 0, stream>>>(qb, kb, vb, relb, attn);
    proj_gemm<<<dim3(4, 1024), 256, SMEM_BYTES, stream>>>(attn, wprjT, proj_b, x, x1);
    ln2_kernel<<<8192, 256, 0, stream>>>(x1, n2g, n2b, h2);
    fc1_gemm<<<dim3(16, 1024), 256, SMEM_BYTES, stream>>>(h2, wf1T, fc1b, fc1o);
    fc2_gemm<<<dim3(4, 1024), 256, SMEM_BYTES, stream>>>(fc1o, wf2T, fc2b, x1, (float*)d_out);
}
